// CausalSelfAttention_27891517620394
// MI455X (gfx1250) — matrix operand order repaired and checked by reading, not compiled, not run on hardware
//
#include <hip/hip_runtime.h>
#include <hip/hip_bf16.h>

#define C_DIM 1024
#define T_LEN 2048
#define B_SZ 4
#define H_NUM 16
#define D_HEAD 64
#define NC3 3072

typedef __attribute__((ext_vector_type(16))) __bf16 v16bf;
typedef __attribute__((ext_vector_type(8)))  float  v8f;

union ABFrag { v16bf v; unsigned short u[16]; };

__device__ __forceinline__ unsigned short f2bf(float f) {
    unsigned int b = __float_as_uint(f);
    b += 0x7FFFu + ((b >> 16) & 1u);   // round-to-nearest-even
    return (unsigned short)(b >> 16);
}

__device__ __forceinline__ void ld8(const unsigned short* p, unsigned short* d) {
    *reinterpret_cast<uint4*>(d) = *reinterpret_cast<const uint4*>(p);
}

__device__ __forceinline__ float rmax16(float v) {
    #pragma unroll
    for (int m = 1; m < 16; m <<= 1) v = fmaxf(v, __shfl_xor(v, m, 32));
    return v;
}
__device__ __forceinline__ float rsum16(float v) {
    #pragma unroll
    for (int m = 1; m < 16; m <<= 1) v += __shfl_xor(v, m, 32);
    return v;
}

#define WMMA_BF16(A, B, Cv) \
    __builtin_amdgcn_wmma_f32_16x16x32_bf16(false, (A), false, (B), (short)0, (Cv), false, false)

// ---------- conversion kernels ----------
__global__ void k_f32_to_bf16(const float* __restrict__ src,
                              unsigned short* __restrict__ dst, int n) {
    int i = blockIdx.x * blockDim.x + threadIdx.x;
    if (i < n) dst[i] = f2bf(src[i]);
}

// W[k][n] (KxN, f32) -> Wt[n][k] (NxK, bf16)
__global__ void k_transpose_bf16(const float* __restrict__ W,
                                 unsigned short* __restrict__ Wt, int K, int N) {
    int i = blockIdx.x * blockDim.x + threadIdx.x;
    if (i < K * N) {
        int n = i / K, k = i % K;
        Wt[(size_t)n * K + k] = f2bf(W[(size_t)k * N + n]);
    }
}

// ---------- shared GEMM fragment helpers (32x64 per wave) ----------
__device__ __forceinline__ void gemm_load(const unsigned short* arow0,
                                          const unsigned short* arow1,
                                          const unsigned short* brow0,
                                          int kk, int aklo, int bklo,
                                          ABFrag* a, ABFrag* b) {
    ld8(arow0 + kk + aklo,      &a[0].u[0]);
    ld8(arow0 + kk + 16 + aklo, &a[0].u[8]);
    ld8(arow1 + kk + aklo,      &a[1].u[0]);
    ld8(arow1 + kk + 16 + aklo, &a[1].u[8]);
    #pragma unroll
    for (int j = 0; j < 4; j++) {
        const unsigned short* br = brow0 + (size_t)(j * 16) * C_DIM + kk;
        ld8(br + bklo,     &b[j].u[0]);
        ld8(br + bklo + 16, &b[j].u[8]);
    }
}

__device__ __forceinline__ void gemm_mma(const ABFrag* a, const ABFrag* b,
                                         v8f acc[2][4]) {
    #pragma unroll
    for (int j = 0; j < 4; j++) {
        acc[0][j] = WMMA_BF16(a[0].v, b[j].v, acc[0][j]);
        acc[1][j] = WMMA_BF16(a[1].v, b[j].v, acc[1][j]);
    }
}

// ---------- QKV GEMM: 32x64 per wave, double-buffered ----------
__global__ void k_qkv_gemm(const unsigned short* __restrict__ xb,
                           const unsigned short* __restrict__ wt,
                           const float* __restrict__ bias,
                           unsigned short* __restrict__ qb,
                           unsigned short* __restrict__ kb,
                           unsigned short* __restrict__ vt) {
    const int lane = threadIdx.x;
    const int ng = blockIdx.x * 4 + threadIdx.y;   // 0..47 (groups of 4 n-tiles)
    const int mt = blockIdx.y;                     // 0..255 (blocks of 32 rows)
    const int l15 = lane & 15;
    const int hi  = lane >> 4;
    const int aklo = hi * 8, bklo = hi * 8;
    const unsigned short* arow0 = xb + (size_t)(mt * 32 + l15) * C_DIM;
    const unsigned short* arow1 = arow0 + (size_t)16 * C_DIM;
    const unsigned short* brow0 = wt + (size_t)(ng * 64 + l15) * C_DIM;

    v8f acc[2][4];
    #pragma unroll
    for (int mi = 0; mi < 2; mi++)
        #pragma unroll
        for (int j = 0; j < 4; j++) acc[mi][j] = (v8f){};

    ABFrag aC[2], bC[4], aN[2], bN[4];
    gemm_load(arow0, arow1, brow0, 0, aklo, bklo, aC, bC);
    for (int kk = 0; kk < C_DIM; kk += 64) {
        gemm_load(arow0, arow1, brow0, kk + 32, aklo, bklo, aN, bN);
        __builtin_prefetch(arow0 + kk + 512, 0, 1);
        gemm_mma(aC, bC, acc);
        if (kk + 64 < C_DIM)
            gemm_load(arow0, arow1, brow0, kk + 64, aklo, bklo, aC, bC);
        gemm_mma(aN, bN, acc);
    }

    #pragma unroll
    for (int j = 0; j < 4; j++) {
        const int cg = ng * 64 + j * 16 + l15;
        const float bv = bias[cg];
        #pragma unroll
        for (int mi = 0; mi < 2; mi++) {
            #pragma unroll
            for (int i = 0; i < 8; i++) {
                float val = acc[mi][j][i] + bv;
                int row = mt * 32 + mi * 16 + i + hi * 8;
                int bz = row >> 11;            // / T_LEN
                int t  = row & (T_LEN - 1);
                if (cg < 1024) {
                    int h = cg >> 6, d = cg & 63;
                    qb[(((size_t)(bz * H_NUM + h)) * T_LEN + t) * D_HEAD + d] = f2bf(val);
                } else if (cg < 2048) {
                    int c2 = cg - 1024, h = c2 >> 6, d = c2 & 63;
                    kb[(((size_t)(bz * H_NUM + h)) * T_LEN + t) * D_HEAD + d] = f2bf(val);
                } else {
                    int c2 = cg - 2048, h = c2 >> 6, d = c2 & 63;
                    vt[(((size_t)(bz * H_NUM + h)) * D_HEAD + d) * T_LEN + t] = f2bf(val);
                }
            }
        }
    }
}

// ---------- flash attention: 32 queries per wave (2 q-tiles share K/V frags) ----------
__global__ void k_attn(const unsigned short* __restrict__ qb,
                       const unsigned short* __restrict__ kb,
                       const unsigned short* __restrict__ vt,
                       unsigned short* __restrict__ ao) {
    __shared__ __align__(16) unsigned short pl[4][32 * 32];
    const int lane = threadIdx.x;
    const int w    = threadIdx.y;
    const int qp = blockIdx.x * 4 + w;   // 0..63, queries [qp*32, qp*32+31]
    const int bh = blockIdx.y;           // 0..63
    const int l15 = lane & 15;
    const int hi  = lane >> 4;
    const int aklo = hi * 8, bklo = hi * 8;
    unsigned short* pw = pl[w];

    // Q A-fragments: [qtile][d-chunk], loaded once
    ABFrag aq[2][2];
    #pragma unroll
    for (int qi = 0; qi < 2; qi++) {
        const unsigned short* qrow =
            qb + (size_t)(bh * T_LEN + qp * 32 + qi * 16 + l15) * D_HEAD;
        ld8(qrow + aklo,      &aq[qi][0].u[0]); ld8(qrow + 16 + aklo, &aq[qi][0].u[8]);
        ld8(qrow + 32 + aklo, &aq[qi][1].u[0]); ld8(qrow + 48 + aklo, &aq[qi][1].u[8]);
    }

    float ms[2][8], ls[2][8];
    v8f o[2][4];
    #pragma unroll
    for (int qi = 0; qi < 2; qi++) {
        #pragma unroll
        for (int i = 0; i < 8; i++) { ms[qi][i] = -1e30f; ls[qi][i] = 0.f; }
        #pragma unroll
        for (int dt = 0; dt < 4; dt++) o[qi][dt] = (v8f){};
    }

    const float scale = 0.125f;  // 1/sqrt(64)
    const unsigned short* vbase = vt + (size_t)bh * D_HEAD * T_LEN;
    const int kend = qp * 32 + 32;

    for (int k0 = 0; k0 < kend; k0 += 32) {
        // K as B-matrix: [key-subtile][d-chunk]
        ABFrag bk[2][2];
        #pragma unroll
        for (int kt = 0; kt < 2; kt++) {
            const unsigned short* kr =
                kb + (size_t)(bh * T_LEN + k0 + kt * 16 + l15) * D_HEAD;
            ld8(kr + bklo,          &bk[kt][0].u[0]); ld8(kr + bklo + 16,      &bk[kt][0].u[8]);
            ld8(kr + 32 + bklo,     &bk[kt][1].u[0]); ld8(kr + 32 + bklo + 16, &bk[kt][1].u[8]);
            __builtin_prefetch(kr + 32 * D_HEAD, 0, 1);   // next 32-key block
        }
        // scores: 2 qtiles x 2 key tiles
        v8f s[2][2];
        #pragma unroll
        for (int qi = 0; qi < 2; qi++)
            #pragma unroll
            for (int kt = 0; kt < 2; kt++) {
                v8f t = {};
                t = WMMA_BF16(aq[qi][0].v, bk[kt][0].v, t);
                t = WMMA_BF16(aq[qi][1].v, bk[kt][1].v, t);
                s[qi][kt] = t;
            }

        // online softmax (row = i + hi*8, col = l15 within each key subtile)
        #pragma unroll
        for (int qi = 0; qi < 2; qi++) {
            #pragma unroll
            for (int i = 0; i < 8; i++) {
                int mrow = i + hi * 8;
                int qg = qp * 32 + qi * 16 + mrow;
                float aS = s[qi][0][i] * scale; if (k0 + l15 > qg)      aS = -1e30f;
                float bS = s[qi][1][i] * scale; if (k0 + 16 + l15 > qg) bS = -1e30f;
                float mx = rmax16(fmaxf(aS, bS));
                float mn = fmaxf(ms[qi][i], mx);
                float corr = __expf(ms[qi][i] - mn);
                float pa = __expf(aS - mn);
                float pb = __expf(bS - mn);
                ls[qi][i] = ls[qi][i] * corr + rsum16(pa + pb);
                ms[qi][i] = mn;
                o[qi][0][i] *= corr; o[qi][1][i] *= corr;
                o[qi][2][i] *= corr; o[qi][3][i] *= corr;
                pw[(qi * 16 + mrow) * 32 + l15]      = f2bf(pa);
                pw[(qi * 16 + mrow) * 32 + 16 + l15] = f2bf(pb);
            }
        }
        // cross-lane LDS handoff: all ds_stores complete before ds_loads
        asm volatile("s_wait_dscnt 0x0" ::: "memory");

        ABFrag pA[2];
        #pragma unroll
        for (int qi = 0; qi < 2; qi++) {
            ld8(&pw[(qi * 16 + l15) * 32 + aklo],      &pA[qi].u[0]);
            ld8(&pw[(qi * 16 + l15) * 32 + 16 + aklo], &pA[qi].u[8]);
        }

        // PV: load each V fragment once, feed both q-tiles
        #pragma unroll
        for (int dt = 0; dt < 4; dt++) {
            ABFrag vB;
            const unsigned short* vr = vbase + (size_t)(dt * 16 + l15) * T_LEN + k0;
            ld8(vr + bklo, &vB.u[0]); ld8(vr + bklo + 16, &vB.u[8]);
            __builtin_prefetch(vr + 32, 0, 1);            // next 32-key block
            o[0][dt] = WMMA_BF16(pA[0].v, vB.v, o[0][dt]);
            o[1][dt] = WMMA_BF16(pA[1].v, vB.v, o[1][dt]);
        }
    }

    const int b = bh >> 4, h = bh & 15;
    #pragma unroll
    for (int qi = 0; qi < 2; qi++) {
        #pragma unroll
        for (int i = 0; i < 8; i++) {
            int t = qp * 32 + qi * 16 + i + hi * 8;
            float inv = 1.f / ls[qi][i];
            size_t rbase = ((size_t)(b * T_LEN + t)) * C_DIM + h * 64 + l15;
            ao[rbase + 0]  = f2bf(o[qi][0][i] * inv);
            ao[rbase + 16] = f2bf(o[qi][1][i] * inv);
            ao[rbase + 32] = f2bf(o[qi][2][i] * inv);
            ao[rbase + 48] = f2bf(o[qi][3][i] * inv);
        }
    }
}

// ---------- projection GEMM: 32x64 per wave, double-buffered -> f32 out ----------
__global__ void k_proj_gemm(const unsigned short* __restrict__ ab,
                            const unsigned short* __restrict__ wt,
                            const float* __restrict__ bias,
                            float* __restrict__ out) {
    const int lane = threadIdx.x;
    const int ng = blockIdx.x * 4 + threadIdx.y;   // 0..15 (groups of 4 n-tiles)
    const int mt = blockIdx.y;                     // 0..255 (blocks of 32 rows)
    const int l15 = lane & 15;
    const int hi  = lane >> 4;
    const int aklo = hi * 8, bklo = hi * 8;
    const unsigned short* arow0 = ab + (size_t)(mt * 32 + l15) * C_DIM;
    const unsigned short* arow1 = arow0 + (size_t)16 * C_DIM;
    const unsigned short* brow0 = wt + (size_t)(ng * 64 + l15) * C_DIM;

    v8f acc[2][4];
    #pragma unroll
    for (int mi = 0; mi < 2; mi++)
        #pragma unroll
        for (int j = 0; j < 4; j++) acc[mi][j] = (v8f){};

    ABFrag aC[2], bC[4], aN[2], bN[4];
    gemm_load(arow0, arow1, brow0, 0, aklo, bklo, aC, bC);
    for (int kk = 0; kk < C_DIM; kk += 64) {
        gemm_load(arow0, arow1, brow0, kk + 32, aklo, bklo, aN, bN);
        __builtin_prefetch(arow0 + kk + 512, 0, 1);
        gemm_mma(aC, bC, acc);
        if (kk + 64 < C_DIM)
            gemm_load(arow0, arow1, brow0, kk + 64, aklo, bklo, aC, bC);
        gemm_mma(aN, bN, acc);
    }

    #pragma unroll
    for (int j = 0; j < 4; j++) {
        const int cg = ng * 64 + j * 16 + l15;
        const float bv = bias[cg];
        #pragma unroll
        for (int mi = 0; mi < 2; mi++) {
            #pragma unroll
            for (int i = 0; i < 8; i++) {
                int row = mt * 32 + mi * 16 + i + hi * 8;
                out[(size_t)row * C_DIM + cg] = acc[mi][j][i] + bv;
            }
        }
    }
}

extern "C" void kernel_launch(void* const* d_in, const int* in_sizes, int n_in,
                              void* d_out, int out_size, void* d_ws, size_t ws_size,
                              hipStream_t stream) {
    (void)in_sizes; (void)n_in; (void)out_size; (void)ws_size;
    const float* x     = (const float*)d_in[0];
    const float* Wqkv  = (const float*)d_in[1];
    const float* bqkv  = (const float*)d_in[2];
    const float* Wproj = (const float*)d_in[3];
    const float* bproj = (const float*)d_in[4];
    float* out = (float*)d_out;

    char* ws = (char*)d_ws;
    size_t off = 0;
    auto carve = [&](size_t bytes) -> void* {
        void* p = ws + off;
        off += (bytes + 255) & ~(size_t)255;
        return p;
    };
    const size_t NTOK = (size_t)B_SZ * T_LEN;            // 8192
    unsigned short* xb     = (unsigned short*)carve(NTOK * C_DIM * 2);
    unsigned short* wqkvt  = (unsigned short*)carve((size_t)NC3 * C_DIM * 2);
    unsigned short* wprojt = (unsigned short*)carve((size_t)C_DIM * C_DIM * 2);
    unsigned short* qb     = (unsigned short*)carve(NTOK * C_DIM * 2);
    unsigned short* kb     = (unsigned short*)carve(NTOK * C_DIM * 2);
    unsigned short* vt     = (unsigned short*)carve(NTOK * C_DIM * 2);
    unsigned short* ao     = (unsigned short*)carve(NTOK * C_DIM * 2);

    const int nx = (int)(NTOK * C_DIM);
    k_f32_to_bf16<<<(nx + 255) / 256, 256, 0, stream>>>(x, xb, nx);
    k_transpose_bf16<<<(NC3 * C_DIM + 255) / 256, 256, 0, stream>>>(Wqkv, wqkvt, C_DIM, NC3);
    k_transpose_bf16<<<(C_DIM * C_DIM + 255) / 256, 256, 0, stream>>>(Wproj, wprojt, C_DIM, C_DIM);

    k_qkv_gemm <<<dim3(12, 256), dim3(32, 4), 0, stream>>>(xb, wqkvt, bqkv, qb, kb, vt);
    k_attn     <<<dim3(16, 64),  dim3(32, 4), 0, stream>>>(qb, kb, vt, ao);
    k_proj_gemm<<<dim3(4, 256),  dim3(32, 4), 0, stream>>>(ao, wprojt, bproj, out);
}
